// GraphEncoder_GAT_3221225472135
// MI455X (gfx1250) — matrix operand order repaired and checked by reading, not compiled, not run on hardware
//
#include <hip/hip_runtime.h>
#include <hip/hip_bf16.h>
#include <hip/hip_fp16.h>

typedef __attribute__((ext_vector_type(16))) _Float16 v16h;
typedef __attribute__((ext_vector_type(8)))  _Float16 v8h;
typedef __attribute__((ext_vector_type(4)))  _Float16 v4h;
typedef __attribute__((ext_vector_type(8)))  float    v8f;

// ---------- helpers ----------
__device__ __forceinline__ unsigned encOrd(float f) {
  unsigned b = __float_as_uint(f);
  return (b & 0x80000000u) ? ~b : (b | 0x80000000u);
}
__device__ __forceinline__ float decOrd(unsigned u) {
  unsigned b = (u & 0x80000000u) ? (u & 0x7fffffffu) : ~u;
  return __uint_as_float(b);
}
__device__ __forceinline__ void atomAddF(float* p, float v) {
  unsafeAtomicAdd(p, v);   // lowers to global_atomic_add_f32 on gfx1250
}

// ---------- generic fill ----------
__global__ void fill_u32_kernel(unsigned* p, unsigned v, long long n) {
  long long i = (long long)blockIdx.x * blockDim.x + threadIdx.x;
  if (i < n) p[i] = v;
}

// ---------- f32 -> f16 bulk convert (4 elems/thread, b128 in / b64 out) ----------
__global__ void f32_to_f16_kernel(const float* __restrict__ src, _Float16* __restrict__ dst,
                                  long long n4) {
  long long i = (long long)blockIdx.x * blockDim.x + threadIdx.x;
  if (i >= n4) return;
  float4 v = ((const float4*)src)[i];
  v4h h; h[0] = (_Float16)v.x; h[1] = (_Float16)v.y; h[2] = (_Float16)v.z; h[3] = (_Float16)v.w;
  ((v4h*)dst)[i] = h;
}

// ---------- pack W[K,Nc] f32 into WMMA B-fragment order (f16) ----------
// Record layout: Wsw[((ct*numKc + kc)*32 + lane)*16 + 2i (+1)] holds
// W[kc*32 + (i>>2)*16 + (lane>>4)*8 + 2*(i&3) (+1), ct*16 + (lane&15)]  (ISA B 32x16 f16 layout)
__global__ void pack_w_kernel(const float* __restrict__ W, _Float16* __restrict__ Wsw,
                              int K, int Nc) {
  int numKc = K >> 5;
  long long total = (long long)(Nc >> 4) * numKc * 256;   // 32 lanes * 8 dwords per group
  long long t = (long long)blockIdx.x * blockDim.x + threadIdx.x;
  if (t >= total) return;
  int i    = (int)(t & 7);
  int lane = (int)((t >> 3) & 31);
  long long grp = t >> 8;                  // = ct*numKc + kc
  int kc = (int)(grp % numKc);
  int ct = (int)(grp / numKc);
  int half = lane >> 4, n = lane & 15;
  int k   = kc * 32 + (i >> 2) * 16 + half * 8 + 2 * (i & 3);
  int col = ct * 16 + n;
  long long base = (grp * 32 + lane) * 16 + 2 * i;
  Wsw[base]     = (_Float16)W[(long long)k * Nc + col];
  Wsw[base + 1] = (_Float16)W[(long long)(k + 1) * Nc + col];
}

// ---------- WMMA GEMM: C[M,Nc] = Ah[M,K] @ W[K,Nc] + bias ----------
// Ah row-major f16, Wsw pre-swizzled f16 fragments. One wave -> 16 rows x 64
// cols (4 col tiles reusing the A fragment). K mult of 32, Nc mult of 64,
// M mult of 16 (all uses satisfy this).
__global__ void wmma_gemm_packed_kernel(const _Float16* __restrict__ Ah,
                                        const _Float16* __restrict__ Wsw,
                                        const float* __restrict__ bias, float* __restrict__ C,
                                        int M, int K, int Nc) {
  int numKc = K >> 5;
  int waveId = threadIdx.x >> 5;
  int lane   = threadIdx.x & 31;
  int rowTile = blockIdx.y * 8 + waveId;
  int row0 = rowTile * 16;
  if (row0 >= M) return;
  int col0 = blockIdx.x * 64;
  int m    = lane & 15;
  int half = lane >> 4;
  const _Float16* Arow = Ah + (long long)(row0 + m) * K;

  v8f acc0 = {}, acc1 = {}, acc2 = {}, acc3 = {};
  for (int kc = 0; kc < numKc; kc++) {
    int k0 = kc << 5;
    // A fragment: two contiguous 8-half runs per lane (ISA A 16x32 f16 layout)
    v8h lo = *(const v8h*)(Arow + k0 + half * 8);
    v8h hi = *(const v8h*)(Arow + k0 + 16 + half * 8);
    v16h a = __builtin_shufflevector(lo, hi, 0, 1, 2, 3, 4, 5, 6, 7,
                                     8, 9, 10, 11, 12, 13, 14, 15);
#pragma unroll
    for (int t = 0; t < 4; t++) {
      long long ct = (col0 >> 4) + t;
      v16h b = *(const v16h*)(Wsw + ((ct * numKc + kc) * 32 + lane) * 16);
      v8f* accp = (t == 0) ? &acc0 : (t == 1) ? &acc1 : (t == 2) ? &acc2 : &acc3;
      *accp = __builtin_amdgcn_wmma_f32_16x16x32_f16(false, a, false, b,
                                                     (short)0, *accp, false, false);
    }
  }
#pragma unroll
  for (int t = 0; t < 4; t++) {
    int col = col0 + t * 16 + m;
    float bv = bias ? bias[col] : 0.f;
    v8f accv = (t == 0) ? acc0 : (t == 1) ? acc1 : (t == 2) ? acc2 : acc3;
#pragma unroll
    for (int j = 0; j < 8; j++) {
      int r = row0 + half * 8 + j;   // C/D layout: VGPR j -> M = half*8 + j
      C[(long long)r * Nc + col] = accv[j] + bv;
    }
  }
}

// ---------- degree + mean edge_attr for self loops ----------
__global__ void deg_loop_kernel(const int* __restrict__ ei, const float* __restrict__ ef,
                                float* __restrict__ loop_acc, float* __restrict__ deg, int E_) {
  long long idx = (long long)blockIdx.x * blockDim.x + threadIdx.x; // E*32
  if (idx >= (long long)E_ * 32) return;
  int e = (int)(idx >> 5), c = (int)(idx & 31);
  int dst = ei[E_ + e];
  atomAddF(&loop_acc[(long long)dst * 32 + c], ef[idx]);
  if (c == 0) atomAddF(&deg[dst], 1.0f);
}
__global__ void loop_div_kernel(float* __restrict__ la, const float* __restrict__ deg, int N_) {
  long long idx = (long long)blockIdx.x * blockDim.x + threadIdx.x; // N*32
  if (idx >= (long long)N_ * 32) return;
  la[idx] /= fmaxf(deg[idx >> 5], 1.0f);
}

// ---------- pass A: alpha_e = <leaky_relu(xl[src]+xr[dst]+ea@We), att> ----------
__global__ void edge_alpha_kernel(const float* __restrict__ xl, const float* __restrict__ xr,
                                  const int* __restrict__ ei, const float* __restrict__ ef,
                                  const float* __restrict__ loop_attr,
                                  const float* __restrict__ We, const float* __restrict__ att,
                                  float* __restrict__ alpha, unsigned* __restrict__ amax,
                                  int E_, int N_) {
  __shared__ float sWe[32 * 256];
  __shared__ float sAtt[256];
  for (int i = threadIdx.x; i < 32 * 256; i += blockDim.x) sWe[i] = We[i];
  for (int i = threadIdx.x; i < 256; i += blockDim.x) sAtt[i] = att[i];
  __syncthreads();

  int wave = threadIdx.x >> 5, lane = threadIdx.x & 31;
  long long e = (long long)blockIdx.x * 8 + wave;
  long long total = (long long)E_ + N_;
  if (e >= total) return;
  int src, dst; const float* ea;
  if (e < E_) { src = ei[e]; dst = ei[E_ + e]; ea = ef + e * 32; }
  else        { src = dst = (int)(e - E_);     ea = loop_attr + (long long)src * 32; }

  float myea = ea[lane];
  int ch = lane * 8;
  const float* pl = xl + (long long)src * 256 + ch;
  const float* pr = xr + (long long)dst * 256 + ch;
  float mv[8];
#pragma unroll
  for (int j = 0; j < 8; j++) mv[j] = pl[j] + pr[j];
  for (int k = 0; k < 32; k++) {
    float ek = __shfl(myea, k, 32);
    const float* w = &sWe[k * 256 + ch];
#pragma unroll
    for (int j = 0; j < 8; j++) mv[j] += ek * w[j];
  }
  float partial = 0.f;
#pragma unroll
  for (int j = 0; j < 8; j++) {
    float v = mv[j];
    v = v > 0.f ? v : 0.2f * v;          // leaky_relu(0.2)
    partial += v * sAtt[ch + j];
  }
  partial += __shfl_xor(partial, 1, 32); // reduce over 8-lane head group
  partial += __shfl_xor(partial, 2, 32);
  partial += __shfl_xor(partial, 4, 32);
  if ((lane & 7) == 0) {
    int head = lane >> 3;
    alpha[e * 4 + head] = partial;
    atomicMax(&amax[(long long)dst * 4 + head], encOrd(partial));
  }
}

// ---------- pass B: a = exp(alpha - amax[dst]); out[dst] += xl[src]*a; denom += a ----------
__global__ void edge_scatter_kernel(const float* __restrict__ xl, const int* __restrict__ ei,
                                    const float* __restrict__ alpha, const unsigned* __restrict__ amax,
                                    float* __restrict__ denom, float* __restrict__ out,
                                    int E_, int N_) {
  int wave = threadIdx.x >> 5, lane = threadIdx.x & 31;
  long long e = (long long)blockIdx.x * 8 + wave;
  long long total = (long long)E_ + N_;
  if (e >= total) return;
  int src, dst;
  if (e < E_) { src = ei[e]; dst = ei[E_ + e]; }
  else        { src = dst = (int)(e - E_); }
  int head = lane >> 3;
  float al = alpha[e * 4 + head];
  float mx = decOrd(amax[(long long)dst * 4 + head]);
  float a  = __expf(al - mx);
  if ((lane & 7) == 0) atomAddF(&denom[(long long)dst * 4 + head], a);
  int ch = lane * 8;
  const float* pl = xl + (long long)src * 256 + ch;
  float* po = out + (long long)dst * 256 + ch;
#pragma unroll
  for (int j = 0; j < 8; j++) atomAddF(&po[j], pl[j] * a);
}

// ---------- finalize: x = out/denom + bias (+relu) (+graph-mean accumulation) ----------
__global__ void gat_finalize_kernel(float* __restrict__ x, const float* __restrict__ denom,
                                    const float* __restrict__ bias, const int* __restrict__ batch,
                                    float* __restrict__ gacc, float* __restrict__ cnt,
                                    int N_, int doRelu, int doBatch) {
  long long idx = (long long)blockIdx.x * blockDim.x + threadIdx.x; // N*256
  if (idx >= (long long)N_ * 256) return;
  int n = (int)(idx >> 8), c = (int)(idx & 255), head = c >> 6;
  float v = x[idx] / denom[(long long)n * 4 + head] + bias[c];
  if (doRelu) v = fmaxf(v, 0.f);
  x[idx] = v;
  if (doBatch) {
    int g = batch[n];
    atomAddF(&gacc[g * 256 + c], v);
    if (c == 0) atomAddF(&cnt[g], 1.0f);
  }
}

__global__ void gmean_div_kernel(float* __restrict__ gacc, const float* __restrict__ cnt) {
  int i = blockIdx.x * blockDim.x + threadIdx.x; // 64*256
  gacc[i] /= fmaxf(cnt[i >> 8], 1.0f);
}

// ---------- LayerNorm over 64 channels, one wave per row ----------
__global__ void layernorm64_kernel(const float* __restrict__ in, const float* __restrict__ g,
                                   const float* __restrict__ b, float* __restrict__ out, int rows) {
  int wave = threadIdx.x >> 5, lane = threadIdx.x & 31;
  long long r = (long long)blockIdx.x * 8 + wave;
  if (r >= rows) return;
  float v0 = in[r * 64 + lane], v1 = in[r * 64 + 32 + lane];
  float s = v0 + v1;
#pragma unroll
  for (int o = 16; o > 0; o >>= 1) s += __shfl_xor(s, o, 32);
  float mu = s * (1.f / 64.f);
  float d0 = v0 - mu, d1 = v1 - mu;
  float q = d0 * d0 + d1 * d1;
#pragma unroll
  for (int o = 16; o > 0; o >>= 1) q += __shfl_xor(q, o, 32);
  float inv = rsqrtf(q * (1.f / 64.f) + 1e-5f);
  out[r * 64 + lane]      = d0 * inv * g[lane]      + b[lane];
  out[r * 64 + 32 + lane] = d1 * inv * g[lane + 32] + b[lane + 32];
}

// ---------- launcher ----------
extern "C" void kernel_launch(void* const* d_in, const int* in_sizes, int n_in,
                              void* d_out, int out_size, void* d_ws, size_t ws_size,
                              hipStream_t stream) {
  const int N = in_sizes[0] / 64;
  const int E = in_sizes[1] / 2;
  const long long EN = (long long)E + N;

  const float* node_feature = (const float*)d_in[0];
  const int*   edge_index   = (const int*)d_in[1];
  const float* edge_feature = (const float*)d_in[2];
  const int*   batch        = (const int*)d_in[3];
  const float* Wl[2]   = {(const float*)d_in[4],  (const float*)d_in[11]};
  const float* bl[2]   = {(const float*)d_in[5],  (const float*)d_in[12]};
  const float* Wr[2]   = {(const float*)d_in[6],  (const float*)d_in[13]};
  const float* br[2]   = {(const float*)d_in[7],  (const float*)d_in[14]};
  const float* We[2]   = {(const float*)d_in[8],  (const float*)d_in[15]};
  const float* att[2]  = {(const float*)d_in[9],  (const float*)d_in[16]};
  const float* bias[2] = {(const float*)d_in[10], (const float*)d_in[17]};
  const float* node_W = (const float*)d_in[18]; const float* node_b = (const float*)d_in[19];
  const float* graph_W = (const float*)d_in[20]; const float* graph_b = (const float*)d_in[21];
  const float* nn_g = (const float*)d_in[22]; const float* nn_b = (const float*)d_in[23];
  const float* gn_g = (const float*)d_in[24]; const float* gn_b = (const float*)d_in[25];
  float* out = (float*)d_out;

  // workspace carve-up
  char* wsp = (char*)d_ws;
  auto carve = [&](long long bytes) { char* p = wsp; wsp += (bytes + 255) & ~255ll; return p; };
  float*     xl        = (float*)carve((long long)N * 256 * 4);
  float*     xr        = (float*)carve((long long)N * 256 * 4);
  float*     xc        = (float*)carve((long long)N * 256 * 4);  // layer in/out accumulator
  _Float16*  Ah        = (_Float16*)carve((long long)N * 256 * 2);
  _Float16*  Wsw       = (_Float16*)carve(256 * 256 * 2);
  float*     alpha     = (float*)carve(EN * 4 * 4);
  unsigned*  amax      = (unsigned*)carve((long long)N * 4 * 4);
  float*     denom     = (float*)carve((long long)N * 4 * 4);
  float*     loop_attr = (float*)carve((long long)N * 32 * 4);
  float*     deg       = (float*)carve((long long)N * 4);
  float*     gacc      = (float*)carve(64 * 256 * 4);
  float*     cnt       = (float*)carve(64 * 4);

  auto fill = [&](void* p, unsigned v, long long n) {
    int blocks = (int)((n + 255) / 256);
    fill_u32_kernel<<<blocks, 256, 0, stream>>>((unsigned*)p, v, n);
  };
  auto convert = [&](const float* src, _Float16* dst, long long n) {
    long long n4 = n >> 2;
    f32_to_f16_kernel<<<(int)((n4 + 255) / 256), 256, 0, stream>>>(src, dst, n4);
  };
  auto gemm = [&](const _Float16* A, const float* W, const float* b, float* C,
                  int M, int K, int Nc) {
    long long packN = (long long)(Nc >> 4) * (K >> 5) * 256;
    pack_w_kernel<<<(int)((packN + 255) / 256), 256, 0, stream>>>(W, Wsw, K, Nc);
    dim3 grid(Nc / 64, (((M + 15) / 16) + 7) / 8);
    wmma_gemm_packed_kernel<<<grid, dim3(256), 0, stream>>>(A, Wsw, b, C, M, K, Nc);
  };
  const unsigned ENC_NEG_INF = 0x007FFFFFu; // encOrd(-inf)
  int edgeBlocks = (int)((EN + 7) / 8);

  // degree + self-loop edge attr (same for both layers)
  fill(deg, 0u, N);
  fill(loop_attr, 0u, (long long)N * 32);
  deg_loop_kernel<<<(int)(((long long)E * 32 + 255) / 256), 256, 0, stream>>>(
      edge_index, edge_feature, loop_attr, deg, E);
  loop_div_kernel<<<(int)(((long long)N * 32 + 255) / 256), 256, 0, stream>>>(loop_attr, deg, N);

  const float* x_in = node_feature;
  int Kin = 64;
  for (int l = 0; l < 2; l++) {
    convert(x_in, Ah, (long long)N * Kin);     // one f16 copy feeds both GEMMs
    gemm(Ah, Wl[l], bl[l], xl, N, Kin, 256);
    gemm(Ah, Wr[l], br[l], xr, N, Kin, 256);
    fill(amax, ENC_NEG_INF, (long long)N * 4);
    fill(denom, 0u, (long long)N * 4);
    fill(xc, 0u, (long long)N * 256);          // safe: prior GEMM reads of xc complete first
    edge_alpha_kernel<<<edgeBlocks, 256, 0, stream>>>(
        xl, xr, edge_index, edge_feature, loop_attr, We[l], att[l], alpha, amax, E, N);
    edge_scatter_kernel<<<edgeBlocks, 256, 0, stream>>>(
        xl, edge_index, alpha, amax, denom, xc, E, N);
    if (l == 1) { fill(gacc, 0u, 64 * 256); fill(cnt, 0u, 64); }
    gat_finalize_kernel<<<(int)(((long long)N * 256 + 255) / 256), 256, 0, stream>>>(
        xc, denom, bias[l], batch, gacc, cnt, N, /*relu=*/(l == 0) ? 1 : 0, /*batch=*/(l == 1) ? 1 : 0);
    x_in = xc;
    Kin = 256;
  }

  // local head: LN(xc @ node_W + node_b) -> out[0 : N*64]
  convert(xc, Ah, (long long)N * 256);
  gemm(Ah, node_W, node_b, xl /*reuse as [N,64]*/, N, 256, 64);
  layernorm64_kernel<<<(N + 7) / 8, 256, 0, stream>>>(xl, nn_g, nn_b, out, N);

  // graph head: LN(gmean @ graph_W + graph_b) -> out[N*64 : ]
  gmean_div_kernel<<<64, 256, 0, stream>>>(gacc, cnt);
  convert(gacc, Ah, 64 * 256);                 // local-head GEMM already consumed Ah
  gemm(Ah, graph_W, graph_b, xr /*reuse as [64,64]*/, 64, 256, 64);
  layernorm64_kernel<<<(64 + 7) / 8, 256, 0, stream>>>(xr, gn_g, gn_b, out + (long long)N * 64, 64);
}
